// VQCodebook_89816356094004
// MI455X (gfx1250) — compile-verified, matrix operand order checked by reading, not run on hardware
//
#include <hip/hip_runtime.h>
#include <hip/hip_bf16.h>
#include <stdint.h>

#define K_CB 8192
#define D_E 256
#define N_VEC 32768
#define EMA 0.99f
#define EPS 1e-5f

typedef __attribute__((ext_vector_type(16))) __bf16 v16bf;
typedef __attribute__((ext_vector_type(8)))  float  v8f;
typedef __attribute__((ext_vector_type(4)))  int    v4i;

#if __has_builtin(__builtin_amdgcn_global_load_async_to_lds_b128)
#define HAVE_ASYNC_LDS 1
typedef __attribute__((address_space(1))) v4i* gb128_t;  // global int4*
typedef __attribute__((address_space(3))) v4i* lb128_t;  // LDS int4*
#else
#define HAVE_ASYNC_LDS 0
#endif

__device__ __forceinline__ unsigned short f2bf_rne(float f) {
  unsigned int u = __float_as_uint(f);
  unsigned int r = 0x7FFFu + ((u >> 16) & 1u);
  return (unsigned short)((u + r) >> 16);
}

// ---------------------------------------------------------------------------
// Kernel 1: codebook prep — ||e||^2 per codeword + bf16 conversion (row-major)
// ---------------------------------------------------------------------------
__global__ __launch_bounds__(256) void vq_prep(const float* __restrict__ emb,
                                               unsigned short* __restrict__ embBf,
                                               float* __restrict__ eNorm) {
  const int wave = threadIdx.x >> 5, lane = threadIdx.x & 31;
  const int k = blockIdx.x * 8 + wave;
  const float* row = emb + (size_t)k * D_E;
  float ss = 0.f;
#pragma unroll
  for (int j = 0; j < 8; ++j) {
    int d = lane + j * 32;
    float v = row[d];
    ss += v * v;
    embBf[(size_t)k * D_E + d] = f2bf_rne(v);
  }
#pragma unroll
  for (int m = 16; m >= 1; m >>= 1) ss += __shfl_xor(ss, m, 32);
  if (lane == 0) eNorm[k] = ss;
}

// ---------------------------------------------------------------------------
// Kernel 2: fused distance GEMM (bf16 WMMA) + running argmin over K=8192
// WG = 256 threads = 8 waves; each wave owns TWO 16-row M-tiles (32 rows);
// WG covers 256 rows. Codebook staged 32 codewords/stage into double-buffered
// LDS via GLOBAL_LOAD_ASYNC_TO_LDS_B128 (ASYNCcnt), overlapped with WMMAs.
// ---------------------------------------------------------------------------
#define LDS_STRIDE 264  // ushorts per codeword: 256 data + 8 pad (528B, bank-spread)
#define N_STAGES (K_CB / 32)

__device__ __forceinline__ void stage_issue(const unsigned short* __restrict__ embBf,
                                            unsigned short* dstLds, int stage, int tid) {
  const int kl = tid >> 3;  // codeword slot 0..31
  const int p  = tid & 7;   // 32-bf16 piece 0..7
  const unsigned short* src = embBf + ((size_t)(stage * 32 + kl) * D_E + p * 32);
  unsigned short* dst = dstLds + kl * LDS_STRIDE + p * 32;
#if HAVE_ASYNC_LDS
  __builtin_amdgcn_global_load_async_to_lds_b128((gb128_t)(src),      (lb128_t)(dst),      0, 0);
  __builtin_amdgcn_global_load_async_to_lds_b128((gb128_t)(src + 8),  (lb128_t)(dst + 8),  0, 0);
  __builtin_amdgcn_global_load_async_to_lds_b128((gb128_t)(src + 16), (lb128_t)(dst + 16), 0, 0);
  __builtin_amdgcn_global_load_async_to_lds_b128((gb128_t)(src + 24), (lb128_t)(dst + 24), 0, 0);
#else
  const uint4* s4 = (const uint4*)src;
  uint4* d4 = (uint4*)dst;
  d4[0] = s4[0]; d4[1] = s4[1]; d4[2] = s4[2]; d4[3] = s4[3];
#endif
}

__device__ __forceinline__ void wait_async_stage() {
#if HAVE_ASYNC_LDS
#if __has_builtin(__builtin_amdgcn_s_wait_asynccnt)
  __builtin_amdgcn_s_wait_asynccnt(0);
#else
  asm volatile("s_wait_asynccnt 0" ::: "memory");
#endif
#endif
}

__global__ __launch_bounds__(256) void vq_argmin(const float* __restrict__ z_e,
                                                 const unsigned short* __restrict__ embBf,
                                                 const float* __restrict__ eNorm,
                                                 int* __restrict__ idxOut) {
  __shared__ __align__(16) unsigned short ldsB[2][32 * LDS_STRIDE];
  const int tid = threadIdx.x;
  const int wave = tid >> 5, lane = tid & 31;
  const int h = lane >> 4, n = lane & 15;
  const int rowBase = blockIdx.x * 256 + wave * 32;

  // --- load two A tiles (rows rowBase..+15 and rowBase+16..+31) once,
  //     converting f32 -> bf16 in the ISA A-matrix layout.
  v16bf A0[8], A1[8];
#pragma unroll
  for (int mt = 0; mt < 2; ++mt) {
    const float* zrow = z_e + (size_t)(rowBase + mt * 16 + n) * D_E;
#pragma unroll
    for (int c = 0; c < 8; ++c) {
      const int g0 = c * 32 + h * 8;       // K = g0..g0+7   -> elems 0..7
      const int g1 = c * 32 + 16 + h * 8;  // K = g1..g1+7   -> elems 8..15
      union { v16bf v; unsigned short s[16]; } ua;
#pragma unroll
      for (int j = 0; j < 8; ++j) {
        ua.s[j]     = f2bf_rne(zrow[g0 + j]);
        ua.s[8 + j] = f2bf_rne(zrow[g1 + j]);
      }
      if (mt == 0) A0[c] = ua.v; else A1[c] = ua.v;
    }
  }

  float bestVal[2][8];
  int   bestIdx[2][8];
#pragma unroll
  for (int mt = 0; mt < 2; ++mt)
#pragma unroll
    for (int r = 0; r < 8; ++r) { bestVal[mt][r] = 3.4e38f; bestIdx[mt][r] = 0; }

  stage_issue(embBf, ldsB[0], 0, tid);  // prologue prefetch

  for (int s = 0; s < N_STAGES; ++s) {
    wait_async_stage();       // this wave's async loads for buf[s&1] landed
    __syncthreads();          // all waves' loads landed / prev compute done
    if (s + 1 < N_STAGES)     // prefetch next stage into the other buffer
      stage_issue(embBf, ldsB[(s + 1) & 1], s + 1, tid);
    const unsigned short* bp = ldsB[s & 1];

#pragma unroll
    for (int t = 0; t < 2; ++t) {
      const int kTile = s * 32 + t * 16;
      const float en = eNorm[kTile + n];  // ||e||^2 for this lane's column
      v8f acc0 = {0.f, 0.f, 0.f, 0.f, 0.f, 0.f, 0.f, 0.f};
      v8f acc1 = {0.f, 0.f, 0.f, 0.f, 0.f, 0.f, 0.f, 0.f};
      union BU { v16bf v; uint4 q[2]; };
      const unsigned short* cb = bp + (t * 16 + n) * LDS_STRIDE + h * 8;
      BU cur, nxt;
      cur.q[0] = *(const uint4*)(cb);
      cur.q[1] = *(const uint4*)(cb + 16);
#pragma unroll
      for (int c = 0; c < 8; ++c) {
        if (c < 7) {  // register double-buffer: hoist next B above the WMMAs
          nxt.q[0] = *(const uint4*)(cb + (c + 1) * 32);
          nxt.q[1] = *(const uint4*)(cb + (c + 1) * 32 + 16);
        }
        acc0 = __builtin_amdgcn_wmma_f32_16x16x32_bf16(
            false, A0[c], false, cur.v, (short)0, acc0, false, false);
        acc1 = __builtin_amdgcn_wmma_f32_16x16x32_bf16(
            false, A1[c], false, cur.v, (short)0, acc1, false, false);
        cur = nxt;
      }
#pragma unroll
      for (int r = 0; r < 8; ++r) {
        float d0 = en - 2.0f * acc0[r];
        float d1 = en - 2.0f * acc1[r];
        if (d0 < bestVal[0][r]) { bestVal[0][r] = d0; bestIdx[0][r] = kTile + n; }
        if (d1 < bestVal[1][r]) { bestVal[1][r] = d1; bestIdx[1][r] = kTile + n; }
      }
    }
  }

  // reduce min+index across the 16 lanes of each half (masks stay in-half)
#pragma unroll
  for (int m = 1; m <= 8; m <<= 1) {
#pragma unroll
    for (int mt = 0; mt < 2; ++mt)
#pragma unroll
      for (int r = 0; r < 8; ++r) {
        float ov = __shfl_xor(bestVal[mt][r], m, 32);
        int   oi = __shfl_xor(bestIdx[mt][r], m, 32);
        if (ov < bestVal[mt][r] || (ov == bestVal[mt][r] && oi < bestIdx[mt][r])) {
          bestVal[mt][r] = ov; bestIdx[mt][r] = oi;
        }
      }
  }
  if (n == 0) {  // lane 0 holds rows 0..7 of each tile, lane 16 rows 8..15
#pragma unroll
    for (int mt = 0; mt < 2; ++mt)
#pragma unroll
      for (int r = 0; r < 8; ++r)
        idxOut[rowBase + mt * 16 + h * 8 + r] = bestIdx[mt][r];
  }
}

// ---------------------------------------------------------------------------
// Kernel 3: gather quantized vectors, e_latent loss partials, segment sums
// ---------------------------------------------------------------------------
__global__ __launch_bounds__(256) void vq_gather_stats(
    const float* __restrict__ z_e, const float* __restrict__ emb,
    const int* __restrict__ idxIn, float* __restrict__ qOut,
    float* __restrict__ idxF, float* __restrict__ counts,
    float* __restrict__ dw, float* __restrict__ lossAcc) {
  const int wave = threadIdx.x >> 5, lane = threadIdx.x & 31;
  const int row = blockIdx.x * 8 + wave;
  const int idx = idxIn[row];
  const float* e = emb + (size_t)idx * D_E;
  const float* z = z_e + (size_t)row * D_E;
  float* q = qOut + (size_t)row * D_E;
  float local = 0.f;
#pragma unroll
  for (int j = 0; j < 8; ++j) {
    int d = lane + j * 32;
    float qv = e[d], zv = z[d];
    q[d] = qv;
    float df = qv - zv;
    local += df * df;
    atomicAdd(&dw[(size_t)idx * D_E + d], zv);
  }
#pragma unroll
  for (int m = 16; m >= 1; m >>= 1) local += __shfl_xor(local, m, 32);
  if (lane == 0) {
    atomicAdd(lossAcc, local);
    atomicAdd(&counts[idx], 1.0f);
    idxF[row] = (float)idx;
  }
}

// ---------------------------------------------------------------------------
// Kernels 4-7: EMA buffer updates + normalized codebook + losses
// ---------------------------------------------------------------------------
__global__ __launch_bounds__(256) void vq_ema_w(const float* __restrict__ ema_w,
                                                const float* __restrict__ dw,
                                                float* __restrict__ out_ema_w) {
  size_t i = (size_t)blockIdx.x * 256 + threadIdx.x;
  out_ema_w[i] = EMA * ema_w[i] + (1.0f - EMA) * dw[i];
}

__global__ __launch_bounds__(256) void vq_ema_cs(const float* __restrict__ ema_cs,
                                                 const float* __restrict__ usage,
                                                 const float* __restrict__ counts,
                                                 float* __restrict__ out_cs,
                                                 float* __restrict__ out_usage,
                                                 float* __restrict__ nAcc) {
  const int k = blockIdx.x * 256 + threadIdx.x;
  float c = counts[k];
  float v = EMA * ema_cs[k] + (1.0f - EMA) * c;
  out_cs[k] = v;
  out_usage[k] = usage[k] + c;
  const int lane = threadIdx.x & 31;
  float vs = v;
#pragma unroll
  for (int m = 16; m >= 1; m >>= 1) vs += __shfl_xor(vs, m, 32);
  if (lane == 0) atomicAdd(nAcc, vs);  // n = sum(new_ema_cs)
}

__global__ __launch_bounds__(256) void vq_new_emb(const float* __restrict__ out_cs,
                                                  const float* __restrict__ out_ema_w,
                                                  const float* __restrict__ nAcc,
                                                  float* __restrict__ out_emb) {
  size_t i = (size_t)blockIdx.x * 256 + threadIdx.x;
  int k = (int)(i >> 8);
  float nsum = *nAcc;
  float cs = (out_cs[k] + EPS) / (nsum + (float)K_CB * EPS) * nsum;
  out_emb[i] = out_ema_w[i] / cs;
}

__global__ void vq_losses(const float* __restrict__ lossAcc,
                          float* __restrict__ lossOut) {
  float el = *lossAcc / (float)((size_t)N_VEC * D_E);
  lossOut[0] = 0.25f * el;  // vq_loss
  lossOut[1] = el;          // e_latent_loss
}

// ---------------------------------------------------------------------------
extern "C" void kernel_launch(void* const* d_in, const int* in_sizes, int n_in,
                              void* d_out, int out_size, void* d_ws, size_t ws_size,
                              hipStream_t stream) {
  const float* z_e    = (const float*)d_in[0];
  const float* emb    = (const float*)d_in[1];
  const float* ecsIn  = (const float*)d_in[2];
  const float* emawIn = (const float*)d_in[3];
  const float* usage  = (const float*)d_in[4];

  char* ws = (char*)d_ws;
  unsigned short* embBf = (unsigned short*)(ws);          // 4 MiB bf16 codebook
  float* eNorm  = (float*)(ws + 4194304);                 // 32 KiB ||e||^2
  int*   idxBuf = (int*)  (ws + 4227072);                 // 128 KiB indices
  float* counts = (float*)(ws + 4358144);                 // 32 KiB segment counts
  float* dw     = (float*)(ws + 4390912);                 // 8 MiB segment sums
  float* accum  = (float*)(ws + 12779520);                // [lossSum, nSum]

  float* out    = (float*)d_out;
  float* qOut   = out;                  // (32,1024,256)
  float* idxF   = out + 8388608;        // (32,1024)
  float* lossO  = out + 8421376;        // vq_loss, e_latent_loss
  float* newEmb = out + 8421378;        // (8192,256)
  float* newCs  = out + 10518530;       // (8192,)
  float* newW   = out + 10526722;       // (8192,256)
  float* newUse = out + 12623874;       // (8192,)

  (void)hipMemsetAsync(ws + 4358144, 0, (size_t)(32768 + 8388608 + 8), stream);

  vq_prep        <<<K_CB / 8,           256, 0, stream>>>(emb, embBf, eNorm);
  vq_argmin      <<<N_VEC / 256,        256, 0, stream>>>(z_e, embBf, eNorm, idxBuf);
  vq_gather_stats<<<N_VEC / 8,          256, 0, stream>>>(z_e, emb, idxBuf, qOut,
                                                          idxF, counts, dw, accum);
  vq_ema_w       <<<(K_CB * D_E) / 256, 256, 0, stream>>>(emawIn, dw, newW);
  vq_ema_cs      <<<K_CB / 256,         256, 0, stream>>>(ecsIn, usage, counts,
                                                          newCs, newUse, accum + 1);
  vq_new_emb     <<<(K_CB * D_E) / 256, 256, 0, stream>>>(newCs, newW, accum + 1, newEmb);
  vq_losses      <<<1, 1, 0, stream>>>(accum, lossO);
}
